// RotaryMemory_89653147337148
// MI455X (gfx1250) — compile-verified, matrix operand order checked by reading, not run on hardware
//
#include <hip/hip_runtime.h>
#include <hip/hip_bf16.h>

// Problem constants (from reference: B=4, T=4096, C=2048)
#define BB   4
#define TT   4096
#define CC   2048
#define HH   1024           // C/2 = complex channels = GEMM N = GEMM K
#define MM   (BB * TT)      // 16384 GEMM rows
#define NCH  32             // chunks along T for the cumprod scan
#define CL   (TT / NCH)     // 128 elements per chunk

typedef __attribute__((ext_vector_type(16))) __bf16 bf16x16;
typedef __attribute__((ext_vector_type(8)))  float  floatx8;
typedef __attribute__((ext_vector_type(4)))  unsigned int u32x4;

union FragU { bf16x16 v; u32x4 u[2]; };

__device__ __forceinline__ float2 cmul(float2 a, float2 b) {
    return make_float2(a.x * b.x - a.y * b.y, a.x * b.y + a.y * b.x);
}

// sigmoid(relu(x))^4
__device__ __forceinline__ float act4(float x) {
    float r  = fmaxf(x, 0.0f);
    float s  = 1.0f / (1.0f + __expf(-r));
    float s2 = s * s;
    return s2 * s2;
}

// ---------------------------------------------------------------------------
// Kernel A: preprocess + per-chunk complex product.
// grid = (HH/256, NCH, BB), block = 256. Lane-contiguous channels -> coalesced.
// ---------------------------------------------------------------------------
__global__ __launch_bounds__(256) void k_chunkprod(const float* __restrict__ x,
                                                   __hip_bfloat16* __restrict__ scaleB,
                                                   float2* __restrict__ chunkp) {
    const int c     = blockIdx.x * 256 + threadIdx.x;   // complex channel
    const int chunk = blockIdx.y;
    const int b     = blockIdx.z;
    const float2* x2 = (const float2*)x;

    float2 prod = make_float2(1.0f, 0.0f);
    const int t0 = chunk * CL;
    #pragma unroll 4
    for (int i = 0; i < CL; ++i) {
        const int t = t0 + i;
        const size_t idx = (size_t)(b * TT + t) * HH + c;
        float2 xv = x2[idx];
        float re = act4(xv.x);
        float im = act4(xv.y);
        float s  = sqrtf(re * re + im * im) + 1e-8f;
        scaleB[idx] = __float2bfloat16(s);
        float inv = 1.0f / s;
        prod = cmul(prod, make_float2(re * inv, im * inv));
    }
    chunkp[(size_t)(b * NCH + chunk) * HH + c] = prod;
}

// ---------------------------------------------------------------------------
// Kernel B: exclusive scan of the 32 chunk carries per (b, c), in place.
// 4096 threads total; loads coalesced across c.
// ---------------------------------------------------------------------------
__global__ __launch_bounds__(256) void k_scan_carries(float2* __restrict__ chunkp) {
    const int idx = blockIdx.x * 256 + threadIdx.x;
    if (idx >= BB * HH) return;
    const int b = idx / HH;
    const int c = idx % HH;
    float2 pre = make_float2(1.0f, 0.0f);
    #pragma unroll
    for (int k = 0; k < NCH; ++k) {
        const size_t off = (size_t)(b * NCH + k) * HH + c;
        float2 t = chunkp[off];
        chunkp[off] = pre;
        pre = cmul(pre, t);
    }
}

// ---------------------------------------------------------------------------
// Kernel C: bf16 WMMA GEMM  proc[m,n] = sum_k scale[m,k] * W[n,k]   (f32 out)
// Per-wave 32(M)x64(N) tile -> 8 accumulators, 12 b128 loads per 8 WMMAs
// (1.5 loads/WMMA, ~21 FLOP/byte of L1/L2 operand traffic).
// 8 waves/block in a 4(M)x2(N) arrangement -> 128x128 block tile.
// Fragment layout per CDNA5 ISA 16-bit 16x32 A/B striping (lane = M%16,
// half = lane/16 selects the K octet; two b128 loads per fragment).
// ---------------------------------------------------------------------------
__device__ __forceinline__ bf16x16 load_frag_bf16(const __hip_bfloat16* __restrict__ rowp,
                                                  int kbase, int half) {
    const char* p = (const char*)(rowp + kbase);
    FragU f;
    f.u[0] = *(const u32x4*)(p + half * 16);        // K = kbase + half*8 .. +7
    f.u[1] = *(const u32x4*)(p + 32 + half * 16);   // K = kbase + 16 + half*8 .. +7
    return f.v;
}

__global__ __launch_bounds__(256) void k_gemm_bf16(const __hip_bfloat16* __restrict__ A,
                                                   const __hip_bfloat16* __restrict__ Wm,
                                                   float* __restrict__ Cp) {
    const int lane = threadIdx.x & 31;
    const int wid  = threadIdx.x >> 5;
    const int wm   = wid & 3;                 // 4 waves along M
    const int wn   = wid >> 2;                // 2 waves along N
    const int m0   = blockIdx.x * 128 + wm * 32;
    const int n0   = blockIdx.y * 128 + wn * 64;
    const int lrow = lane & 15;
    const int half = lane >> 4;

    const __hip_bfloat16* a0p = A  + (size_t)(m0 + lrow)      * HH;
    const __hip_bfloat16* a1p = A  + (size_t)(m0 + 16 + lrow) * HH;
    const __hip_bfloat16* b0p = Wm + (size_t)(n0 + lrow)      * HH;
    const __hip_bfloat16* b1p = Wm + (size_t)(n0 + 16 + lrow) * HH;
    const __hip_bfloat16* b2p = Wm + (size_t)(n0 + 32 + lrow) * HH;
    const __hip_bfloat16* b3p = Wm + (size_t)(n0 + 48 + lrow) * HH;

    floatx8 acc00 = {}, acc01 = {}, acc02 = {}, acc03 = {};
    floatx8 acc10 = {}, acc11 = {}, acc12 = {}, acc13 = {};

    for (int k = 0; k < HH; k += 32) {
        bf16x16 a0 = load_frag_bf16(a0p, k, half);
        bf16x16 a1 = load_frag_bf16(a1p, k, half);
        bf16x16 b0 = load_frag_bf16(b0p, k, half);
        bf16x16 b1 = load_frag_bf16(b1p, k, half);
        bf16x16 b2 = load_frag_bf16(b2p, k, half);
        bf16x16 b3 = load_frag_bf16(b3p, k, half);
        acc00 = __builtin_amdgcn_wmma_f32_16x16x32_bf16(false, a0, false, b0, (short)0, acc00, false, false);
        acc01 = __builtin_amdgcn_wmma_f32_16x16x32_bf16(false, a0, false, b1, (short)0, acc01, false, false);
        acc02 = __builtin_amdgcn_wmma_f32_16x16x32_bf16(false, a0, false, b2, (short)0, acc02, false, false);
        acc03 = __builtin_amdgcn_wmma_f32_16x16x32_bf16(false, a0, false, b3, (short)0, acc03, false, false);
        acc10 = __builtin_amdgcn_wmma_f32_16x16x32_bf16(false, a1, false, b0, (short)0, acc10, false, false);
        acc11 = __builtin_amdgcn_wmma_f32_16x16x32_bf16(false, a1, false, b1, (short)0, acc11, false, false);
        acc12 = __builtin_amdgcn_wmma_f32_16x16x32_bf16(false, a1, false, b2, (short)0, acc12, false, false);
        acc13 = __builtin_amdgcn_wmma_f32_16x16x32_bf16(false, a1, false, b3, (short)0, acc13, false, false);
    }

    // C/D layout: VGPR r -> M = r + 8*half, N = lane & 15
    const int col0 = n0 + lrow;
    #pragma unroll
    for (int r = 0; r < 8; ++r) {
        const size_t row0 = (size_t)(m0 + r + 8 * half) * HH;
        const size_t row1 = row0 + (size_t)16 * HH;
        Cp[row0 + col0]      = acc00[r];
        Cp[row0 + col0 + 16] = acc01[r];
        Cp[row0 + col0 + 32] = acc02[r];
        Cp[row0 + col0 + 48] = acc03[r];
        Cp[row1 + col0]      = acc10[r];
        Cp[row1 + col0 + 16] = acc11[r];
        Cp[row1 + col0 + 32] = acc12[r];
        Cp[row1 + col0 + 48] = acc13[r];
    }
}

// ---------------------------------------------------------------------------
// Kernel D: recompute local cumprod with carry prefix, multiply by proc,
// write interleaved (real, imag) -> out[B,T,C]  (float2 per complex channel).
// ---------------------------------------------------------------------------
__global__ __launch_bounds__(256) void k_finalize(const float* __restrict__ x,
                                                  const float2* __restrict__ chunkp,
                                                  const float* __restrict__ proc,
                                                  float* __restrict__ out) {
    const int c     = blockIdx.x * 256 + threadIdx.x;
    const int chunk = blockIdx.y;
    const int b     = blockIdx.z;
    const float2* x2 = (const float2*)x;
    float2* o2 = (float2*)out;

    float2 prod = chunkp[(size_t)(b * NCH + chunk) * HH + c];  // exclusive carry
    const int t0 = chunk * CL;
    #pragma unroll 4
    for (int i = 0; i < CL; ++i) {
        const int t = t0 + i;
        const size_t idx = (size_t)(b * TT + t) * HH + c;
        float2 xv = x2[idx];
        float re = act4(xv.x);
        float im = act4(xv.y);
        float s  = sqrtf(re * re + im * im) + 1e-8f;
        float inv = 1.0f / s;
        prod = cmul(prod, make_float2(re * inv, im * inv));
        float p = proc[idx];
        o2[idx] = make_float2(prod.x * p, prod.y * p);
    }
}

// ---------------------------------------------------------------------------
extern "C" void kernel_launch(void* const* d_in, const int* in_sizes, int n_in,
                              void* d_out, int out_size, void* d_ws, size_t ws_size,
                              hipStream_t stream) {
    const float*          x  = (const float*)d_in[0];
    const __hip_bfloat16* Wm = (const __hip_bfloat16*)d_in[1];
    float*                out = (float*)d_out;

    char* ws = (char*)d_ws;
    __hip_bfloat16* scaleB = (__hip_bfloat16*)ws;                       // 33,554,432 B
    float2*         chunkp = (float2*)(ws + (size_t)33554432);          //  1,048,576 B
    float*          proc   = (float*)(ws + (size_t)33554432 + 1048576); // 67,108,864 B

    dim3 gridA(HH / 256, NCH, BB);              // (4, 32, 4)
    k_chunkprod<<<gridA, 256, 0, stream>>>(x, scaleB, chunkp);

    k_scan_carries<<<(BB * HH + 255) / 256, 256, 0, stream>>>(chunkp);

    dim3 gridG(MM / 128, HH / 128);             // (128, 8)
    k_gemm_bf16<<<gridG, 256, 0, stream>>>(scaleB, Wm, proc);

    k_finalize<<<gridA, 256, 0, stream>>>(x, chunkp, proc, out);
}
